// ATKT_22522808500586
// MI455X (gfx1250) — matrix operand order repaired and checked by reading, not compiled, not run on hardware
//
#include <hip/hip_runtime.h>
#include <hip/hip_bf16.h>

// ---------- types / WMMA helpers (CDNA5 gfx1250, wave32) ----------
typedef __attribute__((ext_vector_type(16))) __bf16 bf16x16;
typedef __attribute__((ext_vector_type(8)))  __bf16 bf16x8;
typedef __attribute__((ext_vector_type(8)))  float  v8f;

__device__ __forceinline__ float sigmoidf_(float x) { return 1.0f / (1.0f + __expf(-x)); }

// A-matrix 16x32 bf16 fragment (row-major source, leading dim ld elements).
// lane<16: M=lane, K={0..7,16..23}; lane>=16: M=lane-16, K={8..15,24..31}
__device__ __forceinline__ bf16x16 load_a16x32(const __bf16* tile, int ld) {
  int lane = threadIdx.x & 31;
  int row  = lane & 15;
  int kh   = (lane >> 4) << 3;           // 0 or 8
  const __bf16* p = tile + row * ld + kh;
  bf16x8 lo = *(const bf16x8*)(p);
  bf16x8 hi = *(const bf16x8*)(p + 16);
  bf16x16 r;
#pragma unroll
  for (int i = 0; i < 8; ++i) { r[i] = lo[i]; r[i + 8] = hi[i]; }
  return r;
}

// B-matrix 32x16 bf16 fragment from weight stored [N,K] row-major (ld = K).
// lane<16: N=lane, K={0..7,16..23}; lane>=16: N=lane-16, K={8..15,24..31}
__device__ __forceinline__ bf16x16 load_b32x16(const __bf16* tile, int ld) {
  int lane = threadIdx.x & 31;
  int col  = lane & 15;
  int kh   = (lane >> 4) << 3;           // 0 or 8
  const __bf16* p = tile + col * ld + kh;
  bf16x8 lo = *(const bf16x8*)(p);
  bf16x8 hi = *(const bf16x8*)(p + 16);
  bf16x16 r;
#pragma unroll
  for (int i = 0; i < 8; ++i) { r[i] = lo[i]; r[i + 8] = hi[i]; }
  return r;
}

__device__ __forceinline__ v8f wmma_bf16(bf16x16 a, bf16x16 b, v8f c) {
  return __builtin_amdgcn_wmma_f32_16x16x32_bf16(false, a, false, b, (short)0, c, false, false);
}

// ---------- tiny prep kernels ----------
__global__ void k_cvt_bf16(const float* __restrict__ s, __bf16* __restrict__ d, int n) {
  int i = blockIdx.x * blockDim.x + threadIdx.x;
  if (i < n) d[i] = (__bf16)s[i];
}

__global__ void k_bias_sum(const float* a, const float* b, float* o, int n) {
  int i = blockIdx.x * blockDim.x + threadIdx.x;
  if (i < n) o[i] = a[i] + b[i];
}

// interaction_emb gather: row = b*T+t; layout [concept|correct] if correct==1 else swapped
__global__ void k_gather(const int* __restrict__ cseq, const int* __restrict__ rseq,
                         const float* __restrict__ embC, const float* __restrict__ embR,
                         __bf16* __restrict__ X) {
  int row = blockIdx.x;                  // b*T + t  (32768 rows)
  int d   = threadIdx.x;                 // 0..255
  int c = cseq[row], r = rseq[row];
  float cv = embC[c * 256 + d];
  float rv = embR[r * 256 + d];
  __bf16* xr = X + (size_t)row * 512;
  if (r == 1) { xr[d] = (__bf16)cv; xr[256 + d] = (__bf16)rv; }
  else        { xr[d] = (__bf16)rv; xr[256 + d] = (__bf16)cv; }
}

// ---------- xg = X[32768,512] @ Wih^T[512,1024] + bias ----------
// grid (2048, 2), block 256 = 8 waves, each wave 4 n-tiles; ping-pong pipelined K loop
__global__ void __launch_bounds__(256) k_gemm_xg(const __bf16* __restrict__ X,
                                                 const __bf16* __restrict__ W,   // [1024,512]
                                                 const float* __restrict__ bias,
                                                 float* __restrict__ out) {
  const int K = 512, N = 1024;
  int wave = threadIdx.x >> 5, lane = threadIdx.x & 31;
  int mt  = blockIdx.x;
  int nt0 = (blockIdx.y * 8 + wave) * 4;
  const __bf16* At = X + (size_t)mt * 16 * K;
  const __bf16* Wt = W + (size_t)nt0 * 16 * K;
  v8f acc[4] = {{}, {}, {}, {}};
  bf16x16 a0, a1, b0[4], b1[4];
  a0 = load_a16x32(At, K);
#pragma unroll
  for (int j = 0; j < 4; ++j) b0[j] = load_b32x16(Wt + (size_t)j * 16 * K, K);
  for (int k = 0; k < K - 64; k += 64) {
    a1 = load_a16x32(At + k + 32, K);
#pragma unroll
    for (int j = 0; j < 4; ++j) b1[j] = load_b32x16(Wt + (size_t)j * 16 * K + k + 32, K);
#pragma unroll
    for (int j = 0; j < 4; ++j) acc[j] = wmma_bf16(a0, b0[j], acc[j]);
    a0 = load_a16x32(At + k + 64, K);
#pragma unroll
    for (int j = 0; j < 4; ++j) b0[j] = load_b32x16(Wt + (size_t)j * 16 * K + k + 64, K);
#pragma unroll
    for (int j = 0; j < 4; ++j) acc[j] = wmma_bf16(a1, b1[j], acc[j]);
  }
  a1 = load_a16x32(At + K - 32, K);
#pragma unroll
  for (int j = 0; j < 4; ++j) b1[j] = load_b32x16(Wt + (size_t)j * 16 * K + K - 32, K);
#pragma unroll
  for (int j = 0; j < 4; ++j) acc[j] = wmma_bf16(a0, b0[j], acc[j]);
#pragma unroll
  for (int j = 0; j < 4; ++j) acc[j] = wmma_bf16(a1, b1[j], acc[j]);
  int col = lane & 15, rb = (lane >> 4) * 8;
  float* orow = out + (size_t)mt * 16 * N;
#pragma unroll
  for (int j = 0; j < 4; ++j) {
    int n = (nt0 + j) * 16 + col;
    float bn = bias[n];
#pragma unroll
    for (int r = 0; r < 8; ++r)
      orow[(size_t)(rb + r) * N + n] = acc[j][r] + bn;
  }
}

// ---------- sequential LSTM: 4 blocks x 16 batch rows, 16 waves each ----------
__global__ void __launch_bounds__(512) k_lstm(const float* __restrict__ xg,     // [B*T,1024]
                                              const __bf16* __restrict__ Whh,   // [1024,256]
                                              __bf16* __restrict__ H,           // [B*T,256]
                                              __bf16* __restrict__ Fin) {       // [B*T,512] right half
  __shared__ __bf16 hsh[16 * 256];       // A-operand source (8KB)
  __shared__ __bf16 gsh[16 * 1024];      // gate preactivations (32KB)
  const int tid = threadIdx.x, lane = tid & 31, wave = tid >> 5;
  const int br0 = blockIdx.x * 16;
  for (int i = tid; i < 16 * 256; i += 512) hsh[i] = (__bf16)0.0f;
  float creg[8];
#pragma unroll
  for (int j = 0; j < 8; ++j) creg[j] = 0.0f;
  __syncthreads();
  const int nt0 = wave * 4;              // 16 waves * 4 tiles = 64 n-tiles (N=1024)
  const int col = lane & 15, rb = (lane >> 4) * 8;
  const __bf16* Wt = Whh + (size_t)nt0 * 16 * 256;
  for (int t = 0; t < 512; ++t) {
    // issue this step's xg gate loads first: latency hidden behind the 32 WMMAs below
    float xgr[4][8];
#pragma unroll
    for (int j = 0; j < 4; ++j) {
      int n = (nt0 + j) * 16 + col;
#pragma unroll
      for (int r = 0; r < 8; ++r)
        xgr[j][r] = xg[((size_t)(br0 + rb + r) * 512 + t) * 1024 + n];
    }
    // h @ Whh^T, ping-pong pipelined over K (8 slices of 32)
    v8f acc[4] = {{}, {}, {}, {}};
    bf16x16 a0, a1, b0[4], b1[4];
    a0 = load_a16x32(hsh, 256);
#pragma unroll
    for (int j = 0; j < 4; ++j) b0[j] = load_b32x16(Wt + (size_t)j * 16 * 256, 256);
#pragma unroll
    for (int k = 0; k < 192; k += 64) {
      a1 = load_a16x32(hsh + k + 32, 256);
#pragma unroll
      for (int j = 0; j < 4; ++j) b1[j] = load_b32x16(Wt + (size_t)j * 16 * 256 + k + 32, 256);
#pragma unroll
      for (int j = 0; j < 4; ++j) acc[j] = wmma_bf16(a0, b0[j], acc[j]);
      a0 = load_a16x32(hsh + k + 64, 256);
#pragma unroll
      for (int j = 0; j < 4; ++j) b0[j] = load_b32x16(Wt + (size_t)j * 16 * 256 + k + 64, 256);
#pragma unroll
      for (int j = 0; j < 4; ++j) acc[j] = wmma_bf16(a1, b1[j], acc[j]);
    }
    a1 = load_a16x32(hsh + 224, 256);
#pragma unroll
    for (int j = 0; j < 4; ++j) b1[j] = load_b32x16(Wt + (size_t)j * 16 * 256 + 224, 256);
#pragma unroll
    for (int j = 0; j < 4; ++j) acc[j] = wmma_bf16(a0, b0[j], acc[j]);
#pragma unroll
    for (int j = 0; j < 4; ++j) acc[j] = wmma_bf16(a1, b1[j], acc[j]);
#pragma unroll
    for (int j = 0; j < 4; ++j) {
      int n = (nt0 + j) * 16 + col;
#pragma unroll
      for (int r = 0; r < 8; ++r)
        gsh[(rb + r) * 1024 + n] = (__bf16)(acc[j][r] + xgr[j][r]);
    }
    __syncthreads();
#pragma unroll
    for (int j = 0; j < 8; ++j) {
      int idx = tid * 8 + j, m = idx >> 8, d = idx & 255;
      float gi = sigmoidf_((float)gsh[m * 1024 + d]);
      float gf = sigmoidf_((float)gsh[m * 1024 + 256 + d]);
      float gg = tanhf((float)gsh[m * 1024 + 512 + d]);
      float go = sigmoidf_((float)gsh[m * 1024 + 768 + d]);
      float c = gf * creg[j] + gi * gg;
      creg[j] = c;
      float h = go * tanhf(c);
      hsh[m * 256 + d] = (__bf16)h;
      size_t grow = (size_t)(br0 + m) * 512 + t;
      H[grow * 256 + d] = (__bf16)h;
      Fin[grow * 512 + 256 + d] = (__bf16)h;   // right half of final features
    }
    __syncthreads();
  }
}

// ---------- score = simW . tanh(mlpW @ h + mlpB), one wave per 16 rows ----------
__global__ void __launch_bounds__(256) k_score(const __bf16* __restrict__ H,
                                               const __bf16* __restrict__ mlpW,  // [256,256]
                                               const float* __restrict__ mlpB,
                                               const float* __restrict__ simW,
                                               float* __restrict__ score) {
  int wave = threadIdx.x >> 5, lane = threadIdx.x & 31;
  int mt = blockIdx.x * 8 + wave;        // 2048 m-tiles
  const __bf16* At = H + (size_t)mt * 16 * 256;
  bf16x16 afr[8];
#pragma unroll
  for (int kk = 0; kk < 8; ++kk) afr[kk] = load_a16x32(At + kk * 32, 256);
  int col = lane & 15, rb = (lane >> 4) * 8;
  float p[8];
#pragma unroll
  for (int r = 0; r < 8; ++r) p[r] = 0.0f;
  for (int nt = 0; nt < 16; ++nt) {
    v8f acc = {};
#pragma unroll
    for (int kk = 0; kk < 8; ++kk) {
      bf16x16 b = load_b32x16(mlpW + (size_t)nt * 16 * 256 + kk * 32, 256);
      acc = wmma_bf16(afr[kk], b, acc);
    }
    int n = nt * 16 + col;
    float sw = simW[n], mb = mlpB[n];
#pragma unroll
    for (int r = 0; r < 8; ++r) p[r] += tanhf(acc[r] + mb) * sw;
  }
#pragma unroll
  for (int mask = 1; mask < 16; mask <<= 1)
#pragma unroll
    for (int r = 0; r < 8; ++r) p[r] += __shfl_xor(p[r], mask, 32);
  if (col == 0) {
#pragma unroll
    for (int r = 0; r < 8; ++r) score[mt * 16 + rb + r] = p[r];
  }
}

// ---------- causal softmax attention as streaming prefix + exclusive cumsum ----------
__global__ void __launch_bounds__(256) k_attn(const float* __restrict__ score,
                                              const __bf16* __restrict__ H,
                                              __bf16* __restrict__ Fin) {       // left half
  __shared__ float warr[512], sarr[512], scal[512];
  int b = blockIdx.x, d = threadIdx.x;   // 64 blocks x 256 threads
  if (d == 0) {
    float m = -1e30f, s = 0.0f;
    for (int t = 0; t < 512; ++t) {
      float sc = score[b * 512 + t];
      float nm = fmaxf(m, sc);
      float scale = __expf(m - nm);
      float w = __expf(sc - nm);
      s = s * scale + w;
      warr[t] = w; sarr[t] = s; scal[t] = scale;
      m = nm;
    }
  }
  __syncthreads();
  float v = 0.0f, excl = 0.0f;
  for (int t = 0; t < 512; ++t) {
    size_t row = (size_t)b * 512 + t;
    float h = (float)H[row * 256 + d];
    v = v * scal[t] + warr[t] * h;
    Fin[row * 512 + d] = (__bf16)excl;   // exclusive cumsum of attn_out
    excl += v / sarr[t];
  }
}

// ---------- out = sigmoid(Fin[32768,512] @ fcW^T[512,1024] + fcB) ----------
__global__ void __launch_bounds__(256) k_fc(const __bf16* __restrict__ Fin,
                                            const __bf16* __restrict__ W,       // [1024,512]
                                            const float* __restrict__ bias,
                                            float* __restrict__ out) {
  const int K = 512, N = 1024;
  int wave = threadIdx.x >> 5, lane = threadIdx.x & 31;
  int mt  = blockIdx.x;
  int nt0 = (blockIdx.y * 8 + wave) * 4;
  const __bf16* At = Fin + (size_t)mt * 16 * K;
  const __bf16* Wt = W + (size_t)nt0 * 16 * K;
  v8f acc[4] = {{}, {}, {}, {}};
  bf16x16 a0, a1, b0[4], b1[4];
  a0 = load_a16x32(At, K);
#pragma unroll
  for (int j = 0; j < 4; ++j) b0[j] = load_b32x16(Wt + (size_t)j * 16 * K, K);
  for (int k = 0; k < K - 64; k += 64) {
    a1 = load_a16x32(At + k + 32, K);
#pragma unroll
    for (int j = 0; j < 4; ++j) b1[j] = load_b32x16(Wt + (size_t)j * 16 * K + k + 32, K);
#pragma unroll
    for (int j = 0; j < 4; ++j) acc[j] = wmma_bf16(a0, b0[j], acc[j]);
    a0 = load_a16x32(At + k + 64, K);
#pragma unroll
    for (int j = 0; j < 4; ++j) b0[j] = load_b32x16(Wt + (size_t)j * 16 * K + k + 64, K);
#pragma unroll
    for (int j = 0; j < 4; ++j) acc[j] = wmma_bf16(a1, b1[j], acc[j]);
  }
  a1 = load_a16x32(At + K - 32, K);
#pragma unroll
  for (int j = 0; j < 4; ++j) b1[j] = load_b32x16(Wt + (size_t)j * 16 * K + K - 32, K);
#pragma unroll
  for (int j = 0; j < 4; ++j) acc[j] = wmma_bf16(a0, b0[j], acc[j]);
#pragma unroll
  for (int j = 0; j < 4; ++j) acc[j] = wmma_bf16(a1, b1[j], acc[j]);
  int col = lane & 15, rb = (lane >> 4) * 8;
  float* orow = out + (size_t)mt * 16 * N;
#pragma unroll
  for (int j = 0; j < 4; ++j) {
    int n = (nt0 + j) * 16 + col;
    float bn = bias[n];
#pragma unroll
    for (int r = 0; r < 8; ++r)
      orow[(size_t)(rb + r) * N + n] = sigmoidf_(acc[j][r] + bn);
  }
}

// ---------- host ----------
extern "C" void kernel_launch(void* const* d_in, const int* in_sizes, int n_in,
                              void* d_out, int out_size, void* d_ws, size_t ws_size,
                              hipStream_t stream) {
  const int*   cseq = (const int*)d_in[0];
  const int*   rseq = (const int*)d_in[1];
  const float* embC = (const float*)d_in[2];
  const float* embR = (const float*)d_in[3];
  const float* Wih  = (const float*)d_in[4];
  const float* Whh  = (const float*)d_in[5];
  const float* bih  = (const float*)d_in[6];
  const float* bhh  = (const float*)d_in[7];
  const float* mlpW = (const float*)d_in[8];
  const float* mlpB = (const float*)d_in[9];
  const float* simW = (const float*)d_in[10];
  const float* fcW  = (const float*)d_in[11];
  const float* fcB  = (const float*)d_in[12];
  float* out = (float*)d_out;

  char* ws = (char*)d_ws;
  size_t off = 0;
  auto alloc = [&](size_t bytes) { void* p = ws + off; off += (bytes + 255) & ~(size_t)255; return p; };
  __bf16* WihB  = (__bf16*)alloc((size_t)1024 * 512 * 2);
  __bf16* WhhB  = (__bf16*)alloc((size_t)1024 * 256 * 2);
  __bf16* mlpWB = (__bf16*)alloc((size_t)256 * 256 * 2);
  __bf16* fcWB  = (__bf16*)alloc((size_t)1024 * 512 * 2);
  float*  bias  = (float*)alloc((size_t)1024 * 4);
  __bf16* X     = (__bf16*)alloc((size_t)32768 * 512 * 2);
  float*  xg    = (float*)alloc((size_t)32768 * 1024 * 4);
  __bf16* H     = (__bf16*)alloc((size_t)32768 * 256 * 2);
  float*  score = (float*)alloc((size_t)32768 * 4);
  __bf16* Fin   = (__bf16*)alloc((size_t)32768 * 512 * 2);

  k_cvt_bf16<<<(1024 * 512 + 255) / 256, 256, 0, stream>>>(Wih, WihB, 1024 * 512);
  k_cvt_bf16<<<(1024 * 256 + 255) / 256, 256, 0, stream>>>(Whh, WhhB, 1024 * 256);
  k_cvt_bf16<<<(256 * 256 + 255) / 256, 256, 0, stream>>>(mlpW, mlpWB, 256 * 256);
  k_cvt_bf16<<<(1024 * 512 + 255) / 256, 256, 0, stream>>>(fcW, fcWB, 1024 * 512);
  k_bias_sum<<<4, 256, 0, stream>>>(bih, bhh, bias, 1024);
  k_gather<<<32768, 256, 0, stream>>>(cseq, rseq, embC, embR, X);
  k_gemm_xg<<<dim3(2048, 2), 256, 0, stream>>>(X, WihB, bias, xg);
  k_lstm<<<4, 512, 0, stream>>>(xg, WhhB, H, Fin);
  k_score<<<256, 256, 0, stream>>>(H, mlpWB, mlpB, simW, score);
  k_attn<<<64, 256, 0, stream>>>(score, H, Fin);
  k_fc<<<dim3(2048, 2), 256, 0, stream>>>(Fin, fcWB, fcB, out);
}
